// Block_16527034155056
// MI455X (gfx1250) — compile-verified, matrix operand order checked by reading, not run on hardware
//
#include <hip/hip_runtime.h>
#include <hip/hip_bf16.h>
#include <cstdint>
#include <cstddef>

#define D_MODEL 512
#define D_FF    1024
#define SEQ_T   4096
#define BATCH   8
#define NHEAD   8
#define HDIM    64
#define WLOC    128
#define WDIL    64

typedef __attribute__((ext_vector_type(16))) _Float16 v16h;
typedef __attribute__((ext_vector_type(8)))  float    v8f;

#if __has_builtin(__builtin_amdgcn_global_load_async_to_lds_b128)
#define ASYNC_LDS 1
#endif

// ---------------------------------------------------------------- WMMA core
__device__ __forceinline__ v8f wmma_f16(v16h a, v16h b, v8f c) {
  // D = A(16x32,f16) * B(32x16,f16) + C(16x16,f32)
  return __builtin_amdgcn_wmma_f32_16x16x32_f16(
      /*neg_a=*/false, a, /*neg_b=*/false, b,
      /*c_mod=*/(short)0, c, /*reuse_a=*/false, /*reuse_b=*/false);
}

// Load a 16x32 f16 fragment from a row-major LDS tile (row stride `ld` halves).
// Lane L (0..15): row=L, holds K 0..7 (v0..3) and 16..23 (v4..7);
// Lane L+16:      row=L, holds K 8..15 and 24..31. (ISA 7.12.2 layout)
__device__ __forceinline__ v16h ld_frag(const _Float16* p, int ld) {
  const int lane = threadIdx.x & 31;
  const int r    = lane & 15;
  const int kb   = (lane & 16) >> 1;      // 0 or 8 halves
  v16h f;
  ((uint4*)&f)[0] = *(const uint4*)(p + r * ld + kb);
  ((uint4*)&f)[1] = *(const uint4*)(p + r * ld + kb + 16);
  return f;
}

// 16-byte copy global -> LDS: async DMA on CDNA5 (ASYNCcnt), else via VGPRs.
typedef int v4i_vec __attribute__((vector_size(16)));
typedef __attribute__((address_space(1))) v4i_vec* gptr_v4i;
typedef __attribute__((address_space(3))) v4i_vec* lptr_v4i;

__device__ __forceinline__ void copy16_g2l(const _Float16* g, _Float16* l) {
#ifdef ASYNC_LDS
  v4i_vec* gv = (v4i_vec*)g;   // generic, const dropped
  v4i_vec* lv = (v4i_vec*)l;   // generic (points into LDS)
  __builtin_amdgcn_global_load_async_to_lds_b128(
      (gptr_v4i)gv, (lptr_v4i)lv, /*offset=*/0, /*cpol=*/0);
#else
  *(uint4*)l = *(const uint4*)g;
#endif
}

__device__ __forceinline__ void wait_async_loads() {
#ifdef ASYNC_LDS
#if __has_builtin(__builtin_amdgcn_s_wait_asynccnt)
  __builtin_amdgcn_s_wait_asynccnt(0);
#else
  asm volatile("s_wait_asynccnt 0x0" ::: "memory");
#endif
#endif
}

__device__ __forceinline__ float gelu_exact(float x) {
  return 0.5f * x * (1.0f + erff(x * 0.70710678118654752f));
}

// ---------------------------------------------------------------- fp32->fp16
__global__ __launch_bounds__(256) void cvt_f16_kernel(
    const float* __restrict__ s, _Float16* __restrict__ d, int n) {
  int i = blockIdx.x * 256 + threadIdx.x;
  if (i < n) d[i] = (_Float16)s[i];
}

// ---------------------------------------------------------------- LayerNorm
__global__ __launch_bounds__(256) void ln_kernel(
    const float* __restrict__ xin, const float* __restrict__ g,
    const float* __restrict__ bta, const unsigned char* __restrict__ pm,
    float* __restrict__ xmask32, _Float16* __restrict__ xnout,
    int maskIn, int maskOut) {
  __shared__ float red[256];
  const int row = blockIdx.x;
  const int tid = threadIdx.x;
  const float* xr = xin + (size_t)row * D_MODEL;
  const float mk = pm[row] ? 1.0f : 0.0f;
  float v0 = xr[tid], v1 = xr[tid + 256];
  if (maskIn) { v0 *= mk; v1 *= mk; }
  if (xmask32) {
    xmask32[(size_t)row * D_MODEL + tid]       = v0;
    xmask32[(size_t)row * D_MODEL + tid + 256] = v1;
  }
  red[tid] = v0 + v1;
  __syncthreads();
  for (int s = 128; s > 0; s >>= 1) {
    if (tid < s) red[tid] += red[tid + s];
    __syncthreads();
  }
  const float mean = red[0] * (1.0f / D_MODEL);
  __syncthreads();
  const float d0 = v0 - mean, d1 = v1 - mean;
  red[tid] = d0 * d0 + d1 * d1;
  __syncthreads();
  for (int s = 128; s > 0; s >>= 1) {
    if (tid < s) red[tid] += red[tid + s];
    __syncthreads();
  }
  const float rstd = rsqrtf(red[0] * (1.0f / D_MODEL) + 1e-5f);
  float o0 = d0 * rstd * g[tid] + bta[tid];
  float o1 = d1 * rstd * g[tid + 256] + bta[tid + 256];
  if (maskOut) { o0 *= mk; o1 *= mk; }
  xnout[(size_t)row * D_MODEL + tid]       = (_Float16)o0;
  xnout[(size_t)row * D_MODEL + tid + 256] = (_Float16)o1;
}

// ---------------------------------------------------------------- GEMM
// C[M,N] = A[M,K](f16) * W[N,K]^T(f16) + bias
// Block tile 128x128, BK=32, 8 waves each computing 32x64 (2x4 WMMA tiles).
// Double-buffered LDS, async global->LDS staging on CDNA5.
// EPI 0: store f16.  EPI 1: GELU, store f16.  EPI 2: (+resid)*mask, store f32.
template <int EPI>
__global__ __launch_bounds__(256) void gemm_kernel(
    const _Float16* __restrict__ A, const _Float16* __restrict__ W,
    const float* __restrict__ bias, _Float16* __restrict__ o16,
    float* __restrict__ o32, const float* __restrict__ resid,
    const unsigned char* __restrict__ pm, int M, int N, int K) {
  __shared__ __align__(16) _Float16 sA[2][128 * 32];
  __shared__ __align__(16) _Float16 sB[2][128 * 32];
  const int tid  = threadIdx.x;
  const int lane = tid & 31;
  const int wave = tid >> 5;       // 8 waves
  const int wm   = wave & 3;       // 4 waves along M (32 rows each)
  const int wn   = wave >> 2;      // 2 waves along N (64 cols each)
  const int bm0  = blockIdx.x * 128;
  const int bn0  = blockIdx.y * 128;

  // per-thread staging map: 2 b128 chunks for A, 2 for W per K-step
  const int rA0 = (tid * 2 + 0) >> 2, cA0 = ((tid * 2 + 0) & 3) * 8;
  const int rA1 = (tid * 2 + 1) >> 2, cA1 = ((tid * 2 + 1) & 3) * 8;

  v8f acc[2][4] = {};

  auto stage = [&](int buf, int kt) {
    copy16_g2l(A + (size_t)(bm0 + rA0) * K + kt + cA0, &sA[buf][rA0 * 32 + cA0]);
    copy16_g2l(A + (size_t)(bm0 + rA1) * K + kt + cA1, &sA[buf][rA1 * 32 + cA1]);
    copy16_g2l(W + (size_t)(bn0 + rA0) * K + kt + cA0, &sB[buf][rA0 * 32 + cA0]);
    copy16_g2l(W + (size_t)(bn0 + rA1) * K + kt + cA1, &sB[buf][rA1 * 32 + cA1]);
  };

  stage(0, 0);
  int buf = 0;
  for (int kt = 0; kt < K; kt += 32) {
    wait_async_loads();
    __syncthreads();
    if (kt + 32 < K) stage(buf ^ 1, kt + 32);

    const _Float16* At = sA[buf];
    const _Float16* Bt = sB[buf];
    v16h a0 = ld_frag(&At[(wm * 32 + 0)  * 32], 32);
    v16h a1 = ld_frag(&At[(wm * 32 + 16) * 32], 32);
#pragma unroll
    for (int j = 0; j < 4; ++j) {
      v16h bb = ld_frag(&Bt[(wn * 64 + j * 16) * 32], 32);
      acc[0][j] = wmma_f16(a0, bb, acc[0][j]);
      acc[1][j] = wmma_f16(a1, bb, acc[1][j]);
    }
    buf ^= 1;
  }

  const int hi8 = (lane & 16) >> 1;
#pragma unroll
  for (int i = 0; i < 2; ++i)
#pragma unroll
    for (int j = 0; j < 4; ++j)
#pragma unroll
      for (int e = 0; e < 8; ++e) {
        const int gm = bm0 + wm * 32 + i * 16 + hi8 + e;
        const int gn = bn0 + wn * 64 + j * 16 + (lane & 15);
        float v = acc[i][j][e] + bias[gn];
        if (EPI == 1) v = gelu_exact(v);
        if (EPI == 2) {
          v = (v + resid[(size_t)gm * N + gn]) * (pm[gm] ? 1.0f : 0.0f);
          o32[(size_t)gm * N + gn] = v;
        } else {
          o16[(size_t)gm * N + gn] = (_Float16)v;
        }
      }
}

// ---------------------------------------------------------------- local attn
// grid: x = B * 32 blocks * 4 query-chunks, y = head.
// 32 queries x 256 keys per workgroup; RoPE applied at staging.
__global__ __launch_bounds__(256) void local_attn_kernel(
    const _Float16* __restrict__ q, const _Float16* __restrict__ k,
    const _Float16* __restrict__ v, const float* __restrict__ cosT,
    const float* __restrict__ sinT, const unsigned char* __restrict__ pm,
    _Float16* __restrict__ out) {
  __shared__ __align__(16) _Float16 sQ[32 * 64];
  __shared__ __align__(16) _Float16 sKV[256 * 64];  // K rows, reused as V^T
  __shared__ __align__(16) _Float16 sP[32 * 256];
  __shared__ float sBias[256];

  const int tid  = threadIdx.x;
  const int lane = tid & 31;
  const int wave = tid >> 5;
  const int h    = blockIdx.y;
  int c          = blockIdx.x;
  const int b    = c >> 7;
  c &= 127;
  const int blk  = c >> 2;   // 0..31
  const int qc   = c & 3;    // query chunk 0..3

  {  // stage Q with RoPE (rope position = 64 + query-in-block)
    const int row = tid >> 3;
    const int d0  = (tid & 7) * 8;
    const int qg  = qc * 32 + row;
    const int t   = blk * WLOC + qg;
    const int rp  = 64 + qg;
    const size_t base = ((size_t)(b * SEQ_T + t)) * D_MODEL + h * HDIM;
#pragma unroll
    for (int u = 0; u < 8; ++u) {
      const int d = d0 + u;
      const float x  = (float)q[base + d];
      const float xo = (float)q[base + (d ^ 32)];
      const float rot = (d < 32) ? -xo : xo;
      sQ[row * 64 + d] =
          (_Float16)(x * cosT[rp * HDIM + d] + rot * sinT[rp * HDIM + d]);
    }
  }
  {  // stage K with RoPE (rope position = key-in-window 0..255) + mask bias
    const int d0 = (tid & 7) * 8;
#pragma unroll
    for (int s = 0; s < 8; ++s) {
      const int jk = (tid >> 3) + s * 32;
      const int p  = blk * WLOC - 64 + jk;
      const bool valid = (p >= 0) && (p < SEQ_T) && pm[b * SEQ_T + p];
      const size_t base =
          valid ? ((size_t)(b * SEQ_T + p)) * D_MODEL + h * HDIM : 0;
#pragma unroll
      for (int u = 0; u < 8; ++u) {
        const int d = d0 + u;
        const float x  = valid ? (float)k[base + d] : 0.0f;
        const float xo = valid ? (float)k[base + (d ^ 32)] : 0.0f;
        const float rot = (d < 32) ? -xo : xo;
        sKV[jk * 64 + d] =
            (_Float16)(x * cosT[jk * HDIM + d] + rot * sinT[jk * HDIM + d]);
      }
      if (d0 == 0) sBias[jk] = valid ? 0.0f : -30000.0f;
    }
  }
  __syncthreads();

  {  // S = Q K^T * 0.125 + bias ; wave: mtile=w&1, 4 ntiles at (w>>1)*4
    const int mt = wave & 1;
    const int nb = (wave >> 1) * 4;
    v8f acc[4] = {};
#pragma unroll
    for (int ks = 0; ks < 2; ++ks) {
      v16h a = ld_frag(&sQ[(mt * 16) * 64 + ks * 32], 64);
#pragma unroll
      for (int j = 0; j < 4; ++j) {
        v16h bb = ld_frag(&sKV[((nb + j) * 16) * 64 + ks * 32], 64);
        acc[j] = wmma_f16(a, bb, acc[j]);
      }
    }
    const int hi8 = (lane & 16) >> 1;
#pragma unroll
    for (int j = 0; j < 4; ++j)
#pragma unroll
      for (int e = 0; e < 8; ++e) {
        const int m   = mt * 16 + hi8 + e;
        const int col = (nb + j) * 16 + (lane & 15);
        sP[m * 256 + col] = (_Float16)(acc[j][e] * 0.125f + sBias[col]);
      }
  }
  __syncthreads();

  {  // overwrite sKV with V^T [d][key]
#pragma unroll 4
    for (int s = 0; s < 64; ++s) {
      const int e  = s * 256 + tid;
      const int d  = e >> 8;
      const int jk = e & 255;
      const int p  = blk * WLOC - 64 + jk;
      const bool valid = (p >= 0) && (p < SEQ_T);
      sKV[d * 256 + jk] =
          valid ? v[((size_t)(b * SEQ_T + p)) * D_MODEL + h * HDIM + d]
                : (_Float16)0.0f;
    }
  }
  {  // softmax: 8 threads per row, 32 rows
    const int row = tid >> 3, sub = tid & 7;
    float vals[32];
    float mx = -1e30f;
#pragma unroll
    for (int c2 = 0; c2 < 32; ++c2) {
      float x = (float)sP[row * 256 + sub + c2 * 8];
      vals[c2] = x;
      mx = fmaxf(mx, x);
    }
#pragma unroll
    for (int o = 1; o < 8; o <<= 1) mx = fmaxf(mx, __shfl_xor(mx, o, 32));
    float sum = 0.0f;
#pragma unroll
    for (int c2 = 0; c2 < 32; ++c2) {
      float p = __expf(vals[c2] - mx);
      vals[c2] = p;
      sum += p;
    }
#pragma unroll
    for (int o = 1; o < 8; o <<= 1) sum += __shfl_xor(sum, o, 32);
    const float inv = 1.0f / sum;
#pragma unroll
    for (int c2 = 0; c2 < 32; ++c2)
      sP[row * 256 + sub + c2 * 8] = (_Float16)(vals[c2] * inv);
  }
  __syncthreads();

  {  // O = P V : M=32 N=64 K=256 ; one 16x16 tile per wave
    const int mt = wave & 1;
    const int nt = wave >> 1;
    v8f acc = {};
#pragma unroll
    for (int ks = 0; ks < 8; ++ks) {
      v16h a  = ld_frag(&sP[(mt * 16) * 256 + ks * 32], 256);
      v16h bb = ld_frag(&sKV[(nt * 16) * 256 + ks * 32], 256);
      acc = wmma_f16(a, bb, acc);
    }
    const int hi8 = (lane & 16) >> 1;
#pragma unroll
    for (int e = 0; e < 8; ++e) {
      const int m  = mt * 16 + hi8 + e;
      const int gn = nt * 16 + (lane & 15);
      const int t  = blk * WLOC + qc * 32 + m;
      out[((size_t)(b * SEQ_T + t)) * D_MODEL + h * HDIM + gn] =
          (_Float16)acc[e];
    }
  }
}

// ---------------------------------------------------------------- global attn
// grid: x = B * WDIL (window offset), y = head.  64x64 attention per WG.
__global__ __launch_bounds__(256) void global_attn_kernel(
    const _Float16* __restrict__ q, const _Float16* __restrict__ k,
    const _Float16* __restrict__ v, const float* __restrict__ cosT,
    const float* __restrict__ sinT, const unsigned char* __restrict__ pm,
    _Float16* __restrict__ out) {
  __shared__ __align__(16) _Float16 sQ[64 * 64];
  __shared__ __align__(16) _Float16 sKV[64 * 64];  // K rows, reused as V^T
  __shared__ __align__(16) _Float16 sP[64 * 64];
  __shared__ float sBias[64];

  const int tid  = threadIdx.x;
  const int lane = tid & 31;
  const int wave = tid >> 5;
  const int h    = blockIdx.y;
  const int b    = blockIdx.x >> 6;
  const int w0   = blockIdx.x & 63;

  {  // stage Q,K with RoPE at original position t = ipos*64 + w0
    const int row = tid >> 2;
    const int d0  = (tid & 3) * 16;
    const int t   = row * WDIL + w0;
    const size_t base = ((size_t)(b * SEQ_T + t)) * D_MODEL + h * HDIM;
#pragma unroll
    for (int u = 0; u < 16; ++u) {
      const int d = d0 + u;
      const float cc = cosT[(size_t)t * HDIM + d];
      const float ss = sinT[(size_t)t * HDIM + d];
      const float xq  = (float)q[base + d];
      const float xqo = (float)q[base + (d ^ 32)];
      const float xk  = (float)k[base + d];
      const float xko = (float)k[base + (d ^ 32)];
      const float rq = (d < 32) ? -xqo : xqo;
      const float rk = (d < 32) ? -xko : xko;
      sQ[row * 64 + d]  = (_Float16)(xq * cc + rq * ss);
      sKV[row * 64 + d] = (_Float16)(xk * cc + rk * ss);
    }
    if (tid < 64) {
      const int tt = tid * WDIL + w0;
      sBias[tid] = pm[b * SEQ_T + tt] ? 0.0f : -30000.0f;
    }
  }
  __syncthreads();

  {  // S = Q K^T * 0.125 + bias ; wave: mt=w&3, ntiles {nb, nb+1}
    const int mt = wave & 3;
    const int nb = (wave >> 2) * 2;
    v8f acc[2] = {};
#pragma unroll
    for (int ks = 0; ks < 2; ++ks) {
      v16h a = ld_frag(&sQ[(mt * 16) * 64 + ks * 32], 64);
#pragma unroll
      for (int j = 0; j < 2; ++j) {
        v16h bb = ld_frag(&sKV[((nb + j) * 16) * 64 + ks * 32], 64);
        acc[j] = wmma_f16(a, bb, acc[j]);
      }
    }
    const int hi8 = (lane & 16) >> 1;
#pragma unroll
    for (int j = 0; j < 2; ++j)
#pragma unroll
      for (int e = 0; e < 8; ++e) {
        const int m   = mt * 16 + hi8 + e;
        const int col = (nb + j) * 16 + (lane & 15);
        sP[m * 64 + col] = (_Float16)(acc[j][e] * 0.125f + sBias[col]);
      }
  }
  __syncthreads();

  {  // overwrite sKV with V^T [d][ipos]
#pragma unroll 4
    for (int s = 0; s < 16; ++s) {
      const int e  = s * 256 + tid;
      const int d  = e >> 6;
      const int ip = e & 63;
      const int t  = ip * WDIL + w0;
      sKV[d * 64 + ip] =
          v[((size_t)(b * SEQ_T + t)) * D_MODEL + h * HDIM + d];
    }
  }
  {  // softmax: 4 threads per row, 64 rows
    const int row = tid >> 2, sub = tid & 3;
    float vals[16];
    float mx = -1e30f;
#pragma unroll
    for (int c2 = 0; c2 < 16; ++c2) {
      float x = (float)sP[row * 64 + sub + c2 * 4];
      vals[c2] = x;
      mx = fmaxf(mx, x);
    }
#pragma unroll
    for (int o = 1; o < 4; o <<= 1) mx = fmaxf(mx, __shfl_xor(mx, o, 32));
    float sum = 0.0f;
#pragma unroll
    for (int c2 = 0; c2 < 16; ++c2) {
      float p = __expf(vals[c2] - mx);
      vals[c2] = p;
      sum += p;
    }
#pragma unroll
    for (int o = 1; o < 4; o <<= 1) sum += __shfl_xor(sum, o, 32);
    const float inv = 1.0f / sum;
#pragma unroll
    for (int c2 = 0; c2 < 16; ++c2)
      sP[row * 64 + sub + c2 * 4] = (_Float16)(vals[c2] * inv);
  }
  __syncthreads();

  {  // O = P V : M=64 N=64 K=64
    const int mt = wave & 3;
    const int nb = (wave >> 2) * 2;
    v8f acc[2] = {};
#pragma unroll
    for (int ks = 0; ks < 2; ++ks) {
      v16h a = ld_frag(&sP[(mt * 16) * 64 + ks * 32], 64);
#pragma unroll
      for (int j = 0; j < 2; ++j) {
        v16h bb = ld_frag(&sKV[((nb + j) * 16) * 64 + ks * 32], 64);
        acc[j] = wmma_f16(a, bb, acc[j]);
      }
    }
    const int hi8 = (lane & 16) >> 1;
#pragma unroll
    for (int j = 0; j < 2; ++j)
#pragma unroll
      for (int e = 0; e < 8; ++e) {
        const int m  = mt * 16 + hi8 + e;   // ipos
        const int gn = (nb + j) * 16 + (lane & 15);
        const int t  = m * WDIL + w0;
        out[((size_t)(b * SEQ_T + t)) * D_MODEL + h * HDIM + gn] =
            (_Float16)acc[j][e];
      }
  }
}

// ---------------------------------------------------------------- launch
extern "C" void kernel_launch(void* const* d_in, const int* in_sizes, int n_in,
                              void* d_out, int out_size, void* d_ws,
                              size_t ws_size, hipStream_t stream) {
  (void)in_sizes; (void)n_in; (void)out_size; (void)ws_size;
  const float*         x    = (const float*)d_in[0];
  const unsigned char* pm   = (const unsigned char*)d_in[1];
  const float*         cosT = (const float*)d_in[2];
  const float*         sinT = (const float*)d_in[3];
  const float* ln1g = (const float*)d_in[4];
  const float* ln1b = (const float*)d_in[5];
  const float* ln2g = (const float*)d_in[6];
  const float* ln2b = (const float*)d_in[7];
  const float* ln3g = (const float*)d_in[8];
  const float* ln3b = (const float*)d_in[9];

  const int M = BATCH * SEQ_T;  // 32768

  // workspace carve-up
  char* ws = (char*)d_ws;
  float*    xbuf = (float*)(ws + 0);                        // M*512 f32
  _Float16* xn   = (_Float16*)(ws + (size_t)67108864);      // M*512 f16 (also attn-out)
  _Float16* q16  = (_Float16*)(ws + (size_t)100663296);
  _Float16* k16  = (_Float16*)(ws + (size_t)134217728);
  _Float16* v16  = (_Float16*)(ws + (size_t)167772160);
  _Float16* h1   = (_Float16*)(ws + (size_t)201326592);     // M*1024 f16
  _Float16* h2   = (_Float16*)(ws + (size_t)268435456);     // M*1024 f16
  _Float16* w16  = (_Float16*)(ws + (size_t)335544320);     // converted weights

  // convert all weights to f16 once (deterministic)
  const int widx[11]   = {10, 12, 14, 16, 18, 20, 22, 24, 26, 28, 30};
  const int wsizes[11] = {262144, 262144, 262144, 262144, 262144, 262144,
                          262144, 262144, 524288, 1048576, 524288};
  _Float16* wptr[11];
  {
    size_t off = 0;
    for (int i = 0; i < 11; ++i) {
      wptr[i] = w16 + off;
      off += (size_t)wsizes[i];
      cvt_f16_kernel<<<(wsizes[i] + 255) / 256, 256, 0, stream>>>(
          (const float*)d_in[widx[i]], wptr[i], wsizes[i]);
    }
  }
  _Float16 *w_lq = wptr[0], *w_lk = wptr[1], *w_lv = wptr[2], *w_lo = wptr[3];
  _Float16 *w_gq = wptr[4], *w_gk = wptr[5], *w_gv = wptr[6], *w_go = wptr[7];
  _Float16 *w_f1 = wptr[8], *w_f2 = wptr[9], *w_f3 = wptr[10];
  const float* b_lq = (const float*)d_in[11];
  const float* b_lk = (const float*)d_in[13];
  const float* b_lv = (const float*)d_in[15];
  const float* b_lo = (const float*)d_in[17];
  const float* b_gq = (const float*)d_in[19];
  const float* b_gk = (const float*)d_in[21];
  const float* b_gv = (const float*)d_in[23];
  const float* b_go = (const float*)d_in[25];
  const float* b_f1 = (const float*)d_in[27];
  const float* b_f2 = (const float*)d_in[29];
  const float* b_f3 = (const float*)d_in[31];

  const dim3 gD (M / 128, D_MODEL / 128);  // (256, 4)
  const dim3 gFF(M / 128, D_FF / 128);     // (256, 8)

  // ---- phase 1: x = x*mask ; xn = LN1(x)*mask
  ln_kernel<<<M, 256, 0, stream>>>(x, ln1g, ln1b, pm, xbuf, xn, 1, 1);
  // ---- local attention
  gemm_kernel<0><<<gD, 256, 0, stream>>>(xn, w_lq, b_lq, q16, nullptr, nullptr, pm, M, D_MODEL, D_MODEL);
  gemm_kernel<0><<<gD, 256, 0, stream>>>(xn, w_lk, b_lk, k16, nullptr, nullptr, pm, M, D_MODEL, D_MODEL);
  gemm_kernel<0><<<gD, 256, 0, stream>>>(xn, w_lv, b_lv, v16, nullptr, nullptr, pm, M, D_MODEL, D_MODEL);
  local_attn_kernel<<<dim3(BATCH * 32 * 4, NHEAD), 256, 0, stream>>>(
      q16, k16, v16, cosT, sinT, pm, xn);
  gemm_kernel<2><<<gD, 256, 0, stream>>>(xn, w_lo, b_lo, nullptr, xbuf, xbuf, pm, M, D_MODEL, D_MODEL);

  // ---- phase 2: global attention
  ln_kernel<<<M, 256, 0, stream>>>(xbuf, ln2g, ln2b, pm, nullptr, xn, 0, 0);
  gemm_kernel<0><<<gD, 256, 0, stream>>>(xn, w_gq, b_gq, q16, nullptr, nullptr, pm, M, D_MODEL, D_MODEL);
  gemm_kernel<0><<<gD, 256, 0, stream>>>(xn, w_gk, b_gk, k16, nullptr, nullptr, pm, M, D_MODEL, D_MODEL);
  gemm_kernel<0><<<gD, 256, 0, stream>>>(xn, w_gv, b_gv, v16, nullptr, nullptr, pm, M, D_MODEL, D_MODEL);
  global_attn_kernel<<<dim3(BATCH * WDIL, NHEAD), 256, 0, stream>>>(
      q16, k16, v16, cosT, sinT, pm, xn);
  gemm_kernel<2><<<gD, 256, 0, stream>>>(xn, w_go, b_go, nullptr, xbuf, xbuf, pm, M, D_MODEL, D_MODEL);

  // ---- phase 3: FFN
  ln_kernel<<<M, 256, 0, stream>>>(xbuf, ln3g, ln3b, pm, nullptr, xn, 0, 0);
  gemm_kernel<1><<<gFF, 256, 0, stream>>>(xn, w_f1, b_f1, h1, nullptr, nullptr, pm, M, D_FF, D_MODEL);
  gemm_kernel<1><<<gFF, 256, 0, stream>>>(h1, w_f2, b_f2, h2, nullptr, nullptr, pm, M, D_FF, D_FF);
  gemm_kernel<2><<<gD, 256, 0, stream>>>(h2, w_f3, b_f3, nullptr, (float*)d_out, xbuf, pm, M, D_MODEL, D_FF);
}
